// SparseConvFCNet_48421461295997
// MI455X (gfx1250) — matrix operand order repaired and checked by reading, not compiled, not run on hardware
//
#include <hip/hip_runtime.h>
#include <stdint.h>

typedef __bf16 bf16_t;
typedef bf16_t v16bf __attribute__((ext_vector_type(16)));
typedef float  v8f   __attribute__((ext_vector_type(8)));
typedef uint32_t v4u __attribute__((ext_vector_type(4)));

#define BN_EPS 1e-4f
#define LDA_STRIDE 40   // halves per sA row (32 + pad)
#define LDB_STRIDE 40   // halves per sB row

// ---------------------------------------------------------------------------
// Weight pack: f32 (Koff, Cin, Cout) -> bf16 (Koff, Cin/32, Cout, 32)
// i.e. per 32-wide K-chunk, channel-major rows of 32 contiguous k values —
// exactly the sB LDS image, so GEMM staging is pure b128 copies.
// ---------------------------------------------------------------------------
__global__ void weight_pack_kernel(const float* __restrict__ src, bf16_t* __restrict__ dst,
                                   int Cin, int Cout, int Koff) {
    int idx = blockIdx.x * blockDim.x + threadIdx.x;
    const int total = Koff * Cin * Cout;
    if (idx >= total) return;
    const int cinChunks = Cin >> 5;
    const int kk    = idx & 31;
    const int ch    = (idx >> 5) % Cout;
    const int chunk = idx / (32 * Cout);       // 0 .. Koff*cinChunks-1
    const int kch   = chunk % cinChunks;
    const int off   = chunk / cinChunks;
    dst[idx] = (bf16_t)src[((size_t)(off * Cin + kch * 32 + kk)) * Cout + ch];
}

// ---------------------------------------------------------------------------
// input conv: 27-tap gather conv, Cin=3 -> Cout=32 (tiny; scalar VALU)
// ---------------------------------------------------------------------------
__global__ void input_conv_kernel(const float* __restrict__ feats, const float* __restrict__ w,
                                  const int* __restrict__ rule, float* __restrict__ out, int N) {
    int idx = blockIdx.x * blockDim.x + threadIdx.x;
    if (idx >= N * 32) return;
    int site = idx >> 5, oc = idx & 31;
    float acc = 0.0f;
    #pragma unroll 1
    for (int o = 0; o < 27; ++o) {
        int r = rule[o * N + site];
        if (r >= 0) {
            const float* f  = feats + (size_t)r * 3;
            const float* wo = w + o * 96 + oc;      // w[(o*3+c)*32 + oc]
            acc += f[0] * wo[0] + f[1] * wo[32] + f[2] * wo[64];
        }
    }
    out[idx] = acc;
}

// ---------------------------------------------------------------------------
// BN statistics: deterministic two-stage reduction
// ---------------------------------------------------------------------------
__global__ __launch_bounds__(256) void bn_partial_kernel(const float* __restrict__ x, int N, int C,
                                                         float* __restrict__ partial) {
    const int b = blockIdx.x, nb = gridDim.x, t = threadIdx.x;
    const int rowsPer = (N + nb - 1) / nb;
    const int r0 = b * rowsPer;
    int r1 = r0 + rowsPer; if (r1 > N) r1 = N;
    float s1[4] = {0.f, 0.f, 0.f, 0.f}, s2[4] = {0.f, 0.f, 0.f, 0.f};
    for (int r = r0; r < r1; ++r) {
        const float* xr = x + (size_t)r * C;
        #pragma unroll
        for (int j = 0; j < 4; ++j) {
            int c = t + (j << 8);
            if (c < C) { float v = xr[c]; s1[j] += v; s2[j] += v * v; }
        }
    }
    #pragma unroll
    for (int j = 0; j < 4; ++j) {
        int c = t + (j << 8);
        if (c < C) {
            partial[(size_t)b * 2 * C + c]     = s1[j];
            partial[(size_t)b * 2 * C + C + c] = s2[j];
        }
    }
}

__global__ void bn_finalize_kernel(const float* __restrict__ partial, int C, int nb, int N,
                                   float* __restrict__ stats) {
    for (int c = threadIdx.x; c < C; c += blockDim.x) {
        float s = 0.f, ss = 0.f;
        for (int b = 0; b < nb; ++b) {
            s  += partial[(size_t)b * 2 * C + c];
            ss += partial[(size_t)b * 2 * C + C + c];
        }
        float mean = s / (float)N;
        float var  = ss / (float)N - mean * mean;
        stats[c]     = mean;
        stats[C + c] = rsqrtf(var + BN_EPS);
    }
}

// ---------------------------------------------------------------------------
// BNReLU -> bf16 conv-input buffer  /  BNReLU f32 in-place (final layer)
// ---------------------------------------------------------------------------
__global__ void bnrelu_bf16_kernel(const float* __restrict__ x, const float* __restrict__ stats,
                                   const float* __restrict__ g, const float* __restrict__ b,
                                   bf16_t* __restrict__ out, int N, int C) {
    int idx = blockIdx.x * blockDim.x + threadIdx.x;
    if (idx >= N * C) return;
    int c = idx % C;
    float v = (x[idx] - stats[c]) * stats[C + c] * g[c] + b[c];
    out[idx] = (bf16_t)fmaxf(v, 0.0f);
}

__global__ void bnrelu_f32_kernel(float* __restrict__ x, const float* __restrict__ stats,
                                  const float* __restrict__ g, const float* __restrict__ b,
                                  int N, int C) {
    int idx = blockIdx.x * blockDim.x + threadIdx.x;
    if (idx >= N * C) return;
    int c = idx % C;
    float v = (x[idx] - stats[c]) * stats[C + c] * g[c] + b[c];
    x[idx] = fmaxf(v, 0.0f);
}

// ---------------------------------------------------------------------------
// Implicit-GEMM gather conv on WMMA (bf16 in, f32 accumulate).
// Block = 256 threads (8 waves), tile = 64 sites x 64 out-channels,
// waves arranged 4 (sites) x 2 (channels); each wave: 16x32 output via
// 2 accumulators -> 2 v_wmma_f32_16x16x32_bf16 per K-chunk sharing one A frag.
// LDS ping-pong double buffering: one barrier per chunk; staging is pure
// b128 traffic (weights pre-packed to the LDS image by weight_pack_kernel).
// ---------------------------------------------------------------------------
__global__ __launch_bounds__(256) void wmma_conv_kernel(
    const bf16_t* __restrict__ A,        // (Nin, Cin) bf16, post-BNReLU
    const bf16_t* __restrict__ W,        // packed (Koff, Cin/32, Cout, 32) bf16
    const int*    __restrict__ rule,     // (Koff, Nout), -1 = inactive
    const float*  __restrict__ residual, // nullable, (Nout, Cout)
    float* __restrict__ out,             // (Nout, Cout)
    int Nout, int Cin, int Cout, int Koff)
{
    __shared__ __align__(16) bf16_t sA[2][64 * LDA_STRIDE];
    __shared__ __align__(16) bf16_t sB[2][64 * LDB_STRIDE];

    const int t    = threadIdx.x;
    const int lane = t & 31;
    const int wid  = t >> 5;
    const int wm   = wid >> 1;   // 0..3 : 16-site subtile
    const int wn   = wid & 1;    // 0..1 : 32-channel subtile
    const int row0 = blockIdx.x * 64;
    const int cb0  = blockIdx.y * 64;

    // staging: 4 threads per row, one b128 (8 halves) each
    const int st_r  = t >> 2;    // 0..63 : A site row / B channel row
    const int st_kg = t & 3;     // halves [8*kg, 8*kg+8)

    // fragment coordinates (contiguous k = 16*(lane/16) + s, same order for A and B)
    const int fr_row  = (wm << 4) + (lane & 15);
    const int fr_kb   = (lane >> 4) << 4;    // 0 or 16
    const int fb_col0 = (wn << 5) + (lane & 15);
    const int fb_kb   = (lane >> 4) << 4;    // 0 or 16

    v8f acc0 = {}, acc1 = {};

    const int cinChunks = Cin >> 5;
    const int nChunks   = Koff * cinChunks;
    const bool bvalid   = (cb0 + st_r) < Cout;

    // staging registers (next chunk)
    v4u areg = {0u, 0u, 0u, 0u};
    v4u breg = {0u, 0u, 0u, 0u};
    int off_n = 0, kch_n = 0;    // (offset, cin-chunk) of next chunk to stage

    auto loadRegs = [&]() {
        // A gather: one 16B load per thread
        const int site = row0 + st_r;
        int r = -1;
        if (site < Nout) r = rule[(size_t)off_n * Nout + site];
        v4u z = {0u, 0u, 0u, 0u};
        areg = z;
        if (r >= 0)
            areg = *reinterpret_cast<const v4u*>(A + (size_t)r * Cin + (kch_n << 5) + (st_kg << 3));
        // B: packed weights, contiguous 16B per thread
        breg = z;
        if (bvalid) {
            const bf16_t* Wt = W + (((size_t)(off_n * cinChunks + kch_n) * Cout + cb0 + st_r) << 5);
            breg = *reinterpret_cast<const v4u*>(Wt + (st_kg << 3));
        }
        // advance chunk counters (no div/mod in hot loop)
        if (++kch_n == cinChunks) { kch_n = 0; ++off_n; }
    };

    auto storeRegs = [&](int buf) {
        *reinterpret_cast<v4u*>(&sA[buf][st_r * LDA_STRIDE + (st_kg << 3)]) = areg;
        *reinterpret_cast<v4u*>(&sB[buf][st_r * LDB_STRIDE + (st_kg << 3)]) = breg;
    };

    // prologue: stage chunk 0
    loadRegs();
    storeRegs(0);
    __syncthreads();

    for (int c = 0; c < nChunks; ++c) {
        const int buf   = c & 1;
        const bool more = (c + 1 < nChunks);
        if (more) loadRegs();              // global loads for chunk c+1 (latency hidden)

        // fragments: contiguous 16B LDS reads
        v16bf va, vb0, vb1;
        v4u* pa = reinterpret_cast<v4u*>(&va);
        pa[0] = *reinterpret_cast<const v4u*>(&sA[buf][fr_row * LDA_STRIDE + fr_kb]);
        pa[1] = *reinterpret_cast<const v4u*>(&sA[buf][fr_row * LDA_STRIDE + fr_kb + 8]);
        v4u* pb0 = reinterpret_cast<v4u*>(&vb0);
        pb0[0] = *reinterpret_cast<const v4u*>(&sB[buf][fb_col0 * LDB_STRIDE + fb_kb]);
        pb0[1] = *reinterpret_cast<const v4u*>(&sB[buf][fb_col0 * LDB_STRIDE + fb_kb + 8]);
        v4u* pb1 = reinterpret_cast<v4u*>(&vb1);
        pb1[0] = *reinterpret_cast<const v4u*>(&sB[buf][(fb_col0 + 16) * LDB_STRIDE + fb_kb]);
        pb1[1] = *reinterpret_cast<const v4u*>(&sB[buf][(fb_col0 + 16) * LDB_STRIDE + fb_kb + 8]);

        acc0 = __builtin_amdgcn_wmma_f32_16x16x32_bf16(false, va, false, vb0,
                                                       (short)0, acc0, false, false);
        acc1 = __builtin_amdgcn_wmma_f32_16x16x32_bf16(false, va, false, vb1,
                                                       (short)0, acc1, false, false);

        if (more) storeRegs(buf ^ 1);      // fill the other buffer
        __syncthreads();                   // single barrier per chunk
    }

    // ---- epilogue: D layout (lanes 0-15: M 0..7, lanes 16-31: M 8..15) ----
    const int mb = (lane >> 4) << 3;
    #pragma unroll
    for (int half = 0; half < 2; ++half) {
        const int col = cb0 + (wn << 5) + (half << 4) + (lane & 15);
        if (col < Cout) {
            #pragma unroll
            for (int r = 0; r < 8; ++r) {
                const int row = row0 + (wm << 4) + mb + r;
                if (row < Nout) {
                    float v = half ? acc1[r] : acc0[r];
                    if (residual) v += residual[(size_t)row * Cout + col];
                    out[(size_t)row * Cout + col] = v;
                }
            }
        }
    }
}

// ---------------------------------------------------------------------------
// Unpool + concat: out[n, 0:C] = x[n], out[n, C:] = child[parents[n]]
// ---------------------------------------------------------------------------
__global__ void concat_kernel(const float* __restrict__ x, const float* __restrict__ child,
                              const int* __restrict__ parents, float* __restrict__ out,
                              int N, int C, int childW) {
    const int Wt = C + childW;
    long long idx = (long long)blockIdx.x * blockDim.x + threadIdx.x;
    if (idx >= (long long)N * Wt) return;
    int row = (int)(idx / Wt);
    int c   = (int)(idx % Wt);
    out[idx] = (c < C) ? x[(size_t)row * C + c]
                       : child[(size_t)parents[row] * childW + (c - C)];
}

// ---------------------------------------------------------------------------
// Host orchestration
// ---------------------------------------------------------------------------
extern "C" void kernel_launch(void* const* d_in, const int* in_sizes, int n_in,
                              void* d_out, int out_size, void* d_ws, size_t ws_size,
                              hipStream_t stream) {
    (void)n_in; (void)out_size; (void)ws_size;

    // input indexing (setup_inputs dict order, pytree-flattened):
    // 0 coords, 1 feats, 2 w_in, 3..44 blocks(7x6), 45..62 downs(6x3),
    // 63 g_out, 64 b_out, 65..71 nbrs, 72..77 down_rules, 78..83 parents
    int C[7], N[7], Wd[7];
    for (int l = 0; l < 7; ++l) C[l] = 32 * (l + 1);
    for (int l = 0; l < 7; ++l) N[l] = in_sizes[65 + l] / 27;
    Wd[6] = C[6];
    for (int l = 5; l >= 0; --l) Wd[l] = C[l] + Wd[l + 1];

    const float* feats = (const float*)d_in[1];
    const float* w_in  = (const float*)d_in[2];
    const float* g_out = (const float*)d_in[63];
    const float* b_out = (const float*)d_in[64];

    // ---- workspace bump allocator ----
    size_t off = 0;
    auto alloc = [&](size_t bytes) -> void* {
        off = (off + 255) & ~(size_t)255;
        void* p = (char*)d_ws + off;
        off += bytes;
        return p;
    };

    bf16_t *w1b[7], *w2b[7], *wdb[6];
    float  *xb[7], *outb[6];
    for (int l = 0; l < 7; ++l) {
        size_t wsz = (size_t)27 * C[l] * C[l];
        w1b[l] = (bf16_t*)alloc(wsz * 2);
        w2b[l] = (bf16_t*)alloc(wsz * 2);
    }
    for (int l = 0; l < 6; ++l)
        wdb[l] = (bf16_t*)alloc((size_t)8 * C[l] * C[l + 1] * 2);
    size_t maxNC = 0;
    for (int l = 0; l < 7; ++l) {
        xb[l] = (float*)alloc((size_t)N[l] * C[l] * 4);
        size_t nc = (size_t)N[l] * C[l];
        if (nc > maxNC) maxNC = nc;
    }
    bf16_t* abuf = (bf16_t*)alloc(maxNC * 2);
    float*  ybuf = (float*)alloc(maxNC * 4);
    for (int l = 1; l < 6; ++l) outb[l] = (float*)alloc((size_t)N[l] * Wd[l] * 4);
    float* partial = (float*)alloc((size_t)64 * 2 * 896 * 4);
    float* stats   = (float*)alloc((size_t)2 * 896 * 4);

    auto cdiv = [](long long a, long long b) { return (unsigned)((a + b - 1) / b); };
    auto wpack = [&](const float* s, bf16_t* d, int Cin, int Cout, int Koff) {
        long long n = (long long)Koff * Cin * Cout;
        weight_pack_kernel<<<cdiv(n, 256), 256, 0, stream>>>(s, d, Cin, Cout, Koff);
    };
    auto bnstats = [&](const float* x, int Nn, int Cc) {
        bn_partial_kernel<<<64, 256, 0, stream>>>(x, Nn, Cc, partial);
        bn_finalize_kernel<<<1, 256, 0, stream>>>(partial, Cc, 64, Nn, stats);
    };
    auto bnrelu16 = [&](const float* x, const float* g, const float* b, bf16_t* d, int Nn, int Cc) {
        bnrelu_bf16_kernel<<<cdiv((long long)Nn * Cc, 256), 256, 0, stream>>>(x, stats, g, b, d, Nn, Cc);
    };
    auto wconv = [&](const bf16_t* A, const bf16_t* Wp, const int* rule, const float* res,
                     float* o, int Nout, int Cin, int Cout, int Koff) {
        dim3 g(cdiv(Nout, 64), cdiv(Cout, 64));
        wmma_conv_kernel<<<g, 256, 0, stream>>>(A, Wp, rule, res, o, Nout, Cin, Cout, Koff);
    };

    // ---- weights: f32 -> bf16, packed to the LDS staging image ----
    for (int l = 0; l < 7; ++l) {
        wpack((const float*)d_in[5 + 6 * l], w1b[l], C[l], C[l], 27);
        wpack((const float*)d_in[8 + 6 * l], w2b[l], C[l], C[l], 27);
    }
    for (int l = 0; l < 6; ++l)
        wpack((const float*)d_in[47 + 3 * l], wdb[l], C[l], C[l + 1], 8);

    // ---- input conv: feats (N0,3) -> x0 (N0,32) ----
    input_conv_kernel<<<cdiv((long long)N[0] * 32, 256), 256, 0, stream>>>(
        feats, w_in, (const int*)d_in[65], xb[0], N[0]);

    // ---- descend: residual block per level, strided down-conv between levels ----
    for (int l = 0; l < 7; ++l) {
        const float* g1 = (const float*)d_in[3 + 6 * l];
        const float* b1 = (const float*)d_in[4 + 6 * l];
        const float* g2 = (const float*)d_in[6 + 6 * l];
        const float* b2 = (const float*)d_in[7 + 6 * l];
        const int* nbr  = (const int*)d_in[65 + l];

        bnstats(xb[l], N[l], C[l]);
        bnrelu16(xb[l], g1, b1, abuf, N[l], C[l]);
        wconv(abuf, w1b[l], nbr, nullptr, ybuf, N[l], C[l], C[l], 27);

        bnstats(ybuf, N[l], C[l]);
        bnrelu16(ybuf, g2, b2, abuf, N[l], C[l]);
        wconv(abuf, w2b[l], nbr, xb[l], xb[l], N[l], C[l], C[l], 27);  // fused residual

        if (l < 6) {
            const float* gd = (const float*)d_in[45 + 3 * l];
            const float* bd = (const float*)d_in[46 + 3 * l];
            bnstats(xb[l], N[l], C[l]);
            bnrelu16(xb[l], gd, bd, abuf, N[l], C[l]);
            wconv(abuf, wdb[l], (const int*)d_in[72 + l], nullptr,
                  xb[l + 1], N[l + 1], C[l], C[l + 1], 8);
        }
    }

    // ---- ascend: unpool + concat (level-0 result goes straight into d_out) ----
    const float* child = xb[6];
    int childW = Wd[6];
    for (int l = 5; l >= 0; --l) {
        float* dst = (l == 0) ? (float*)d_out : outb[l];
        concat_kernel<<<cdiv((long long)N[l] * Wd[l], 256), 256, 0, stream>>>(
            xb[l], child, (const int*)d_in[78 + l], dst, N[l], C[l], childW);
        child = dst;
        childW = Wd[l];
    }

    // ---- final BNReLU over 896 channels, in place on d_out ----
    bnstats((const float*)d_out, N[0], Wd[0]);
    bnrelu_f32_kernel<<<cdiv((long long)N[0] * Wd[0], 256), 256, 0, stream>>>(
        (float*)d_out, stats, g_out, b_out, N[0], Wd[0]);
}
